// GAT_Net_65515431133294
// MI455X (gfx1250) — compile-verified, matrix operand order checked by reading, not run on hardware
//
#include <hip/hip_runtime.h>
#include <hip/hip_bf16.h>

// ---------------------------------------------------------------------------
// gfx1250 WMMA (wave32):  D(16x16 f32) = A(16x32 f16) * B(32x16 f16) + C
// ---------------------------------------------------------------------------
typedef __attribute__((ext_vector_type(16))) _Float16 v16h;
typedef __attribute__((ext_vector_type(8)))  _Float16 v8h;
typedef __attribute__((ext_vector_type(8)))  float    v8f;

__device__ __forceinline__ v8f zero8() {
  v8f z;
#pragma unroll
  for (int i = 0; i < 8; ++i) z[i] = 0.f;
  return z;
}

__device__ __forceinline__ v8f wmma_f16(v16h a, v16h b, v8f c) {
  // (neg_a, A, neg_b, B, c_mod, C, reuse_a, reuse_b)
  return __builtin_amdgcn_wmma_f32_16x16x32_f16(false, a, false, b, (short)0, c,
                                                false, false);
}

// Fragment from f16 memory laid out so the two 8-element k-groups are at
// p[0..7] and p[16..23] (16B-aligned): two b128 loads + static shuffle.
__device__ __forceinline__ v16h frag_f16(const _Float16* p) {
  const v8h lo = *(const v8h*)(p);
  const v8h hi = *(const v8h*)(p + 16);
  return __builtin_shufflevector(lo, hi, 0, 1, 2, 3, 4, 5, 6, 7,
                                 8, 9, 10, 11, 12, 13, 14, 15);
}

// A-fragment built from row-major f32 global memory (same k-group layout),
// converted to f16 with constant-index packing.
__device__ __forceinline__ v16h a_frag_f32(const float* __restrict__ p) {
  const float4 x0 = *(const float4*)(p);
  const float4 x1 = *(const float4*)(p + 4);
  const float4 y0 = *(const float4*)(p + 16);
  const float4 y1 = *(const float4*)(p + 20);
  v16h a;
  a[0]  = (_Float16)x0.x; a[1]  = (_Float16)x0.y; a[2]  = (_Float16)x0.z; a[3]  = (_Float16)x0.w;
  a[4]  = (_Float16)x1.x; a[5]  = (_Float16)x1.y; a[6]  = (_Float16)x1.z; a[7]  = (_Float16)x1.w;
  a[8]  = (_Float16)y0.x; a[9]  = (_Float16)y0.y; a[10] = (_Float16)y0.z; a[11] = (_Float16)y0.w;
  a[12] = (_Float16)y1.x; a[13] = (_Float16)y1.y; a[14] = (_Float16)y1.z; a[15] = (_Float16)y1.w;
  return a;
}

__device__ __forceinline__ float sigf(float x) { return 1.f / (1.f + __expf(-x)); }

// ---------------------------------------------------------------------------
// C = act(A @ B + bias).  Compile-time: N (and ldc == N), lda == 64, K == 64.
// bTrans==1: B stored [n][k] (ldb = k-stride); bTrans==0: B stored [k][n].
// One wave per 16-row tile, 8 waves / block. B staged in LDS as [n][k] f16.
// ---------------------------------------------------------------------------
template <int N, bool HAS_BIAS, bool ACT>
__global__ void __launch_bounds__(256)
gemm_wmma(const float* __restrict__ A, const float* __restrict__ B, int ldb,
          int bTrans, const float* __restrict__ bias, float* __restrict__ C,
          int M) {
  __shared__ _Float16 sB[N * 64];  // [n][k], K = 64
  const int tid = threadIdx.x;
  if (bTrans) {
    for (int idx = tid; idx < N * 64; idx += 256)
      sB[idx] = (_Float16)B[(size_t)(idx >> 6) * ldb + (idx & 63)];
  } else {
    for (int idx = tid; idx < N * 64; idx += 256)
      sB[idx] = (_Float16)B[(size_t)(idx & 63) * ldb + (idx >> 6)];
  }
  __syncthreads();

  const int lane = tid & 31, wv = tid >> 5;
  const int row0 = (blockIdx.x * 8 + wv) * 16;
  if (row0 >= M) return;

  const int ln = lane & 15, h8 = (lane >> 4) << 3;
  const float* ap = A + (size_t)(row0 + ln) * 64 + h8;
  const v16h a0 = a_frag_f32(ap);
  const v16h a1 = a_frag_f32(ap + 32);

#pragma unroll
  for (int nt = 0; nt < (N >> 4); ++nt) {
    const _Float16* bp = sB + ((nt * 16 + ln) << 6) + h8;
    v8f acc = zero8();
    acc = wmma_f16(a0, frag_f16(bp), acc);
    acc = wmma_f16(a1, frag_f16(bp + 32), acc);
    const int col = nt * 16 + ln;
    float bv = 0.f;
    if (HAS_BIAS) bv = bias[col];
    float* cp = C + (size_t)(row0 + h8) * N + col;
#pragma unroll
    for (int r = 0; r < 8; ++r) {
      float v = acc[r] + bv;
      if (ACT) v = fmaxf(v, 0.f);
      cp[r * N] = v;
    }
  }
}

// ---------------------------------------------------------------------------
// GRU scan (one direction). One wave owns 16 batch rows through all 15 steps.
// gi_t(row bn, col g) = U[bn][g] + V[b*16 + j_idx(i,t)][g]   (U includes bih)
// gh = h @ Whh^T (WMMA, h staged as f16 in per-wave LDS), + bhh elementwise.
// Also emits logit partials: part[bn][t][c] = h_t . W_hard[dirOff:dirOff+64, c]
// ---------------------------------------------------------------------------
__global__ void __launch_bounds__(256)
gru_scan_wmma(const float* __restrict__ U, const float* __restrict__ V,
              const float* __restrict__ Whh, const float* __restrict__ bhh,
              const float* __restrict__ Whard, int dirOff, int backward,
              float* __restrict__ part) {
  __shared__ _Float16 sW[192 * 64];  // [n][k] == Whh row-major
  __shared__ float    sBh[192];
  __shared__ float    sWh[64 * 2];
  __shared__ _Float16 sH[8][16 * 64];

  const int tid = threadIdx.x;
  for (int idx = tid; idx < 192 * 64; idx += 256) sW[idx] = (_Float16)Whh[idx];
  for (int idx = tid; idx < 192; idx += 256) sBh[idx] = bhh[idx];
  for (int idx = tid; idx < 128; idx += 256) sWh[idx] = Whard[dirOff * 2 + idx];

  const int wv = tid >> 5, lane = tid & 31;
  for (int idx = lane; idx < 16 * 64; idx += 32) sH[wv][idx] = (_Float16)0.f;
  __syncthreads();

  const int bn0 = (blockIdx.x * 8 + wv) * 16;
  const int ln = lane & 15, h8 = (lane >> 4) << 3;

  v8f hF[4];
#pragma unroll
  for (int nt = 0; nt < 4; ++nt) hF[nt] = zero8();

  for (int s = 0; s < 15; ++s) {
    const int t = backward ? (14 - s) : s;

    // Re-pack h (f16, per-wave LDS) into A fragments: 4x ds_load_b128
    const _Float16* hp = &sH[wv][(ln << 6) + h8];
    const v16h a0 = frag_f16(hp);
    const v16h a1 = frag_f16(hp + 32);

    // gh = h @ Whh^T : 12 output tiles (192 cols), K = 64
    v8f gF[12];
#pragma unroll
    for (int g = 0; g < 12; ++g) {
      const _Float16* bp = sW + ((g * 16 + ln) << 6) + h8;
      v8f acc = zero8();
      acc = wmma_f16(a0, frag_f16(bp), acc);
      acc = wmma_f16(a1, frag_f16(bp + 32), acc);
      gF[g] = acc;
    }

    // Elementwise GRU update, write h_new to LDS (f16) for next step
#pragma unroll
    for (int r = 0; r < 8; ++r) {
      const int m = r + h8;
      const int bn = bn0 + m;
      const int i = bn & 15;
      const int j = (t < i) ? t : t + 1;
      const size_t ur = (size_t)bn * 192;
      const size_t vr = (size_t)(((bn >> 4) << 4) | j) * 192;
#pragma unroll
      for (int nt = 0; nt < 4; ++nt) {
        const int c = nt * 16 + ln;
        const float gir = U[ur + c]       + V[vr + c];
        const float giz = U[ur + 64 + c]  + V[vr + 64 + c];
        const float gin = U[ur + 128 + c] + V[vr + 128 + c];
        const float ghr = gF[nt][r]     + sBh[c];
        const float ghz = gF[nt + 4][r] + sBh[64 + c];
        const float ghn = gF[nt + 8][r] + sBh[128 + c];
        const float rg = sigf(gir + ghr);
        const float zg = sigf(giz + ghz);
        const float ng = tanhf(gin + rg * ghn);
        const float hn = (1.f - zg) * ng + zg * hF[nt][r];
        hF[nt][r] = hn;
        sH[wv][(m << 6) + c] = (_Float16)hn;
      }
    }

    // Fused logit partial: lane -> (row m2 = lane>>1, class c = lane&1)
    {
      const int m2 = lane >> 1, cc = lane & 1;
      float acc = 0.f;
#pragma unroll 8
      for (int n = 0; n < 64; ++n)
        acc += (float)sH[wv][(m2 << 6) + n] * sWh[n * 2 + cc];
      part[((size_t)(bn0 + m2) * 15 + t) * 2 + cc] = acc;
    }
    __syncthreads();
  }
}

// ---------------------------------------------------------------------------
// Gumbel hard gate: hard_w = softmax2((logits + g) / tau)[:, 1]
// ---------------------------------------------------------------------------
__global__ void __launch_bounds__(256)
hardw_kernel(const float* __restrict__ pf, const float* __restrict__ pb,
             const float* __restrict__ gum, const float* __restrict__ b_hard,
             float* __restrict__ hw, int total) {
  const int idx = blockIdx.x * 256 + threadIdx.x;
  if (idx >= total) return;
  const float l0 = pf[idx * 2]     + pb[idx * 2]     + b_hard[0];
  const float l1 = pf[idx * 2 + 1] + pb[idx * 2 + 1] + b_hard[1];
  const float u0 = gum[idx * 2], u1 = gum[idx * 2 + 1];
  const float g0 = -__logf(-__logf(u0 + 1e-10f) + 1e-10f);
  const float g1 = -__logf(-__logf(u1 + 1e-10f) + 1e-10f);
  const float a0 = (l0 + g0) * 100.f, a1 = (l1 + g1) * 100.f;  // 1/tau = 100
  const float mx = fmaxf(a0, a1);
  const float e0 = __expf(a0 - mx), e1 = __expf(a1 - mx);
  hw[idx] = e1 / (e0 + e1);
}

// ---------------------------------------------------------------------------
// Attention: one wave per (b,i). scores over 15 neighbors, softmax, gate, sum v.
// ---------------------------------------------------------------------------
__global__ void __launch_bounds__(256)
attn_kernel(const float* __restrict__ q, const float* __restrict__ k,
            const float* __restrict__ v, const float* __restrict__ hw,
            float* __restrict__ x) {
  __shared__ float sWgt[8][16];
  const int wv = threadIdx.x >> 5, lane = threadIdx.x & 31;
  const int bn = blockIdx.x * 8 + wv;
  const int b = bn >> 4, i = bn & 15;

  float sc = -1e30f;
  if (lane < 15) {
    const int j = (lane < i) ? lane : lane + 1;
    const float* qp = q + (size_t)bn * 64;
    const float* kp = k + (size_t)((b << 4) | j) * 64;
    float d = 0.f;
#pragma unroll 8
    for (int a = 0; a < 64; ++a) d += qp[a] * kp[a];
    sc = d * 0.125f;  // 1/sqrt(64)
  }
  float mx = sc;
#pragma unroll
  for (int o = 8; o >= 1; o >>= 1) mx = fmaxf(mx, __shfl_xor(mx, o, 16));
  const float e = (lane < 15) ? __expf(sc - mx) : 0.f;
  float ssum = e;
#pragma unroll
  for (int o = 8; o >= 1; o >>= 1) ssum += __shfl_xor(ssum, o, 16);
  const float wgt = (lane < 15) ? (e / ssum) * hw[(size_t)bn * 15 + lane] : 0.f;
  if (lane < 16) sWgt[wv][lane] = wgt;
  __syncthreads();

  float acc0 = 0.f, acc1 = 0.f;
  for (int l = 0; l < 15; ++l) {
    const int j = (l < i) ? l : l + 1;
    const float* vp = v + (size_t)((b << 4) | j) * 64;
    const float wl = sWgt[wv][l];
    acc0 += wl * vp[lane];
    acc1 += wl * vp[lane + 32];
  }
  x[(size_t)bn * 64 + lane] = acc0;
  x[(size_t)bn * 64 + lane + 32] = acc1;
}

// ---------------------------------------------------------------------------
// Final GRU cell: out = GRUCell(x, h) with Wih_c/Whh_c (192x64 each).
// ---------------------------------------------------------------------------
__global__ void __launch_bounds__(256)
gru_cell_wmma(const float* __restrict__ x, const float* __restrict__ h,
              const float* __restrict__ Wih, const float* __restrict__ Whh,
              const float* __restrict__ bih, const float* __restrict__ bhh,
              float* __restrict__ out) {
  __shared__ _Float16 sWi[192 * 64];
  __shared__ _Float16 sWr[192 * 64];
  __shared__ float sbi[192], sbh[192];
  const int tid = threadIdx.x;
  for (int idx = tid; idx < 192 * 64; idx += 256) {
    sWi[idx] = (_Float16)Wih[idx];
    sWr[idx] = (_Float16)Whh[idx];
  }
  for (int idx = tid; idx < 192; idx += 256) { sbi[idx] = bih[idx]; sbh[idx] = bhh[idx]; }
  __syncthreads();

  const int lane = tid & 31, wv = tid >> 5;
  const int row0 = (blockIdx.x * 8 + wv) * 16;
  const int ln = lane & 15, h8 = (lane >> 4) << 3;

  const float* xp = x + (size_t)(row0 + ln) * 64 + h8;
  const float* hp = h + (size_t)(row0 + ln) * 64 + h8;
  const v16h aX0 = a_frag_f32(xp), aX1 = a_frag_f32(xp + 32);
  const v16h aH0 = a_frag_f32(hp), aH1 = a_frag_f32(hp + 32);

#pragma unroll
  for (int nt = 0; nt < 4; ++nt) {
    const _Float16* biR = sWi + (((nt) * 16 + ln) << 6) + h8;
    const _Float16* biZ = sWi + (((nt + 4) * 16 + ln) << 6) + h8;
    const _Float16* biN = sWi + (((nt + 8) * 16 + ln) << 6) + h8;
    const _Float16* brR = sWr + (((nt) * 16 + ln) << 6) + h8;
    const _Float16* brZ = sWr + (((nt + 4) * 16 + ln) << 6) + h8;
    const _Float16* brN = sWr + (((nt + 8) * 16 + ln) << 6) + h8;
    v8f giR = zero8(), giZ = zero8(), giN = zero8();
    v8f ghR = zero8(), ghZ = zero8(), ghN = zero8();
    giR = wmma_f16(aX0, frag_f16(biR), giR); giR = wmma_f16(aX1, frag_f16(biR + 32), giR);
    giZ = wmma_f16(aX0, frag_f16(biZ), giZ); giZ = wmma_f16(aX1, frag_f16(biZ + 32), giZ);
    giN = wmma_f16(aX0, frag_f16(biN), giN); giN = wmma_f16(aX1, frag_f16(biN + 32), giN);
    ghR = wmma_f16(aH0, frag_f16(brR), ghR); ghR = wmma_f16(aH1, frag_f16(brR + 32), ghR);
    ghZ = wmma_f16(aH0, frag_f16(brZ), ghZ); ghZ = wmma_f16(aH1, frag_f16(brZ + 32), ghZ);
    ghN = wmma_f16(aH0, frag_f16(brN), ghN); ghN = wmma_f16(aH1, frag_f16(brN + 32), ghN);

    const int col = nt * 16 + ln;
#pragma unroll
    for (int r = 0; r < 8; ++r) {
      const size_t row = (size_t)(row0 + r + h8);
      const float hv = h[row * 64 + col];
      const float rg = sigf(giR[r] + sbi[col] + ghR[r] + sbh[col]);
      const float zg = sigf(giZ[r] + sbi[64 + col] + ghZ[r] + sbh[64 + col]);
      const float ng = tanhf(giN[r] + sbi[128 + col] + rg * (ghN[r] + sbh[128 + col]));
      out[row * 64 + col] = (1.f - zg) * ng + zg * hv;
    }
  }
}

// ---------------------------------------------------------------------------
extern "C" void kernel_launch(void* const* d_in, const int* in_sizes, int n_in,
                              void* d_out, int out_size, void* d_ws, size_t ws_size,
                              hipStream_t stream) {
  (void)in_sizes; (void)n_in; (void)out_size; (void)ws_size;
  const float* obs    = (const float*)d_in[0];
  const float* hidden = (const float*)d_in[1];
  const float* gumbel = (const float*)d_in[2];
  const float* W_enc  = (const float*)d_in[3];
  const float* b_enc  = (const float*)d_in[4];
  const float* Wih_f  = (const float*)d_in[5];
  const float* Whh_f  = (const float*)d_in[6];
  const float* bih_f  = (const float*)d_in[7];
  const float* bhh_f  = (const float*)d_in[8];
  const float* Wih_b  = (const float*)d_in[9];
  const float* Whh_b  = (const float*)d_in[10];
  const float* bih_b  = (const float*)d_in[11];
  const float* bhh_b  = (const float*)d_in[12];
  const float* W_hard = (const float*)d_in[13];
  const float* b_hard = (const float*)d_in[14];
  const float* Wq     = (const float*)d_in[15];
  const float* Wk     = (const float*)d_in[16];
  const float* Wv     = (const float*)d_in[17];
  const float* b_v    = (const float*)d_in[18];
  const float* Wih_c  = (const float*)d_in[19];
  const float* Whh_c  = (const float*)d_in[20];
  const float* bih_c  = (const float*)d_in[21];
  const float* bhh_c  = (const float*)d_in[22];

  const int BN = 16384;  // B*N
  float* ws   = (float*)d_ws;
  float* hEnc = ws;                ws += (size_t)BN * 64;
  float* q    = ws;                ws += (size_t)BN * 64;
  float* k    = ws;                ws += (size_t)BN * 64;
  float* v    = ws;                ws += (size_t)BN * 64;
  float* Uf   = ws;                ws += (size_t)BN * 192;
  float* Vf   = ws;                ws += (size_t)BN * 192;
  float* Ub   = ws;                ws += (size_t)BN * 192;
  float* Vb   = ws;                ws += (size_t)BN * 192;
  float* pf   = ws;                ws += (size_t)BN * 15 * 2;
  float* pb   = ws;                ws += (size_t)BN * 15 * 2;
  float* hw   = ws;                ws += (size_t)BN * 15;
  float* xbuf = ws;                ws += (size_t)BN * 64;

  const dim3 b256(256);
  const dim3 gTile(128);  // 128 blocks * 8 waves * 16 rows = 16384

  // Encoder + projections (all K = 64)
  gemm_wmma<64, true, true><<<gTile, b256, 0, stream>>>(obs, W_enc, 64, 0, b_enc, hEnc, BN);
  gemm_wmma<64, false, false><<<gTile, b256, 0, stream>>>(hEnc, Wq, 64, 0, nullptr, q, BN);
  gemm_wmma<64, false, false><<<gTile, b256, 0, stream>>>(hEnc, Wk, 64, 0, nullptr, k, BN);
  gemm_wmma<64, true, true><<<gTile, b256, 0, stream>>>(hEnc, Wv, 64, 0, b_v, v, BN);

  // Factorized GRU inputs: U = hEnc @ Wih[:, :64]^T + bih ; V = hEnc @ Wih[:, 64:]^T
  gemm_wmma<192, true, false><<<gTile, b256, 0, stream>>>(hEnc, Wih_f, 128, 1, bih_f, Uf, BN);
  gemm_wmma<192, false, false><<<gTile, b256, 0, stream>>>(hEnc, Wih_f + 64, 128, 1, nullptr, Vf, BN);
  gemm_wmma<192, true, false><<<gTile, b256, 0, stream>>>(hEnc, Wih_b, 128, 1, bih_b, Ub, BN);
  gemm_wmma<192, false, false><<<gTile, b256, 0, stream>>>(hEnc, Wih_b + 64, 128, 1, nullptr, Vb, BN);

  // Bidirectional GRU scan with fused logit partials
  gru_scan_wmma<<<gTile, b256, 0, stream>>>(Uf, Vf, Whh_f, bhh_f, W_hard, 0, 0, pf);
  gru_scan_wmma<<<gTile, b256, 0, stream>>>(Ub, Vb, Whh_b, bhh_b, W_hard, 64, 1, pb);

  // Gumbel-softmax hard gate
  hardw_kernel<<<dim3((BN * 15 + 255) / 256), b256, 0, stream>>>(pf, pb, gumbel, b_hard, hw,
                                                                 BN * 15);

  // Gated attention -> x
  attn_kernel<<<dim3(BN / 8), b256, 0, stream>>>(q, k, v, hw, xbuf);

  // Final GRU cell -> output
  gru_cell_wmma<<<gTile, b256, 0, stream>>>(xbuf, hidden, Wih_c, Whh_c, bih_c, bhh_c,
                                            (float*)d_out);
}
